// MambaClassifier_14018773254853
// MI455X (gfx1250) — matrix operand order repaired and checked by reading, not compiled, not run on hardware
//
#include <hip/hip_runtime.h>
#include <hip/hip_bf16.h>

#define D_MODEL_ 768
#define NC_ 10
#define N_LAYERS_ 2
#define N_STATE_ 16
#define K_CONV_ 4
#define D_IN_ 1536
#define DT_RANK_ 48
#define B_SZ_ 2
#define L_SEQ_ 2048
#define M_TOK_ (B_SZ_ * L_SEQ_)   // 4096
#define XDBL_W_ (DT_RANK_ + 2 * N_STATE_)  // 80

typedef __attribute__((ext_vector_type(16))) __bf16 v16bf;
typedef __attribute__((ext_vector_type(8)))  __bf16 vbf8;
typedef __attribute__((ext_vector_type(8)))  float  v8f;

// ---------------------------------------------------------------------------
// f32 -> bf16 conversion pass (weights / staged activations)
// ---------------------------------------------------------------------------
__global__ void mamba_cvt_bf16(const float* __restrict__ src, __bf16* __restrict__ dst,
                               long n) {
  long tid = (long)blockIdx.x * blockDim.x + threadIdx.x;
  if (tid < n) dst[tid] = (__bf16)src[tid];
}

// ---------------------------------------------------------------------------
// Generic GEMM on bf16 operands: Out[m,n] = act( sum_k Act[m,k]*W[n,k] + bias[n] )
// One wave -> 16 rows x 64 cols. Inner loop: pure b128/b64 loads + WMMA,
// zero conversions. Optional bf16 mirror of the output for the next GEMM.
// act_mode: 0 = identity, 1 = softplus
// ---------------------------------------------------------------------------
__global__ __launch_bounds__(128)
void mamba_gemm(const __bf16* __restrict__ Act, const __bf16* __restrict__ W,
                const float* __restrict__ bias, float* __restrict__ Out,
                __bf16* __restrict__ Outbf,
                int M, int K, int lda, int N, int act_mode) {
  const int lane = threadIdx.x & 31;
  const int wid  = threadIdx.x >> 5;
  const int half = lane >> 4;        // which 16-lane half of the wave
  const int l16  = lane & 15;
  const int m0 = blockIdx.y * 64 + wid * 16;
  const int n0 = blockIdx.x * 64;
  const int mrow = m0 + l16;
  const long arow = (long)min(mrow, M - 1) * (long)lda;

  long wrow[4];
  #pragma unroll
  for (int j = 0; j < 4; ++j)
    wrow[j] = (long)min(n0 + 16 * j + l16, N - 1) * (long)K;

  const v8f vz = {0.f, 0.f, 0.f, 0.f, 0.f, 0.f, 0.f, 0.f};
  v8f acc[4];
  #pragma unroll
  for (int j = 0; j < 4; ++j) acc[j] = vz;

  // ---- main loop: full 32-wide K slabs, no divergence, no conversions
  int kb = 0;
  for (; kb + 32 <= K; kb += 32) {
    // A operand 16x32 bf16: lane half 0: K={0..15},
    //                       lane half 1: K={16..31}
    const v16bf a = *(const v16bf*)(Act + arow + kb + 16 * half);

    // B operand 32x16 bf16: lane holds 16 consecutive K of weight row n
    #pragma unroll
    for (int j = 0; j < 4; ++j) {
      const v16bf bv = *(const v16bf*)(W + wrow[j] + kb + 16 * half);
      acc[j] = __builtin_amdgcn_wmma_f32_16x16x32_bf16(
          false, a, false, bv, (short)0, acc[j], false, false);
    }
  }

  // ---- tail: K not a multiple of 32 (dt_proj K=48), zero-padded
  if (kb < K) {
    const __bf16 zbf = (__bf16)0.0f;
    v16bf a;
    #pragma unroll
    for (int e = 0; e < 8; ++e) {
      const int k0 = kb + 16 * half + e;
      const int k1 = kb + 16 * half + 8 + e;
      a[e]     = (k0 < K) ? Act[arow + k0] : zbf;
      a[8 + e] = (k1 < K) ? Act[arow + k1] : zbf;
    }
    #pragma unroll
    for (int j = 0; j < 4; ++j) {
      v16bf bv;
      #pragma unroll
      for (int e = 0; e < 16; ++e) {
        const int k = kb + 16 * half + e;
        bv[e] = (k < K) ? W[wrow[j] + k] : zbf;
      }
      acc[j] = __builtin_amdgcn_wmma_f32_16x16x32_bf16(
          false, a, false, bv, (short)0, acc[j], false, false);
    }
  }

  // ---- store: VGPR v -> row m0 + v + 8*half, col = n0 + 16j + l16
  #pragma unroll
  for (int j = 0; j < 4; ++j) {
    const int col = n0 + 16 * j + l16;
    if (col >= N) continue;
    const float bval = bias ? bias[col] : 0.f;
    #pragma unroll
    for (int v = 0; v < 8; ++v) {
      const int row = m0 + v + 8 * half;
      if (row >= M) continue;
      float val = acc[j][v] + bval;
      if (act_mode == 1) val = (val > 20.f) ? val : log1pf(__expf(val));  // softplus
      Out[(long)row * N + col] = val;
      if (Outbf) Outbf[(long)row * N + col] = (__bf16)val;
    }
  }
}

// ---------------------------------------------------------------------------
__global__ void mamba_embed(const int* __restrict__ tokens,
                            const float* __restrict__ emb, __bf16* __restrict__ hbf) {
  long tid = (long)blockIdx.x * blockDim.x + threadIdx.x;
  if (tid >= (long)M_TOK_ * D_MODEL_) return;
  int tok = tokens[tid / D_MODEL_];
  hbf[tid] = (__bf16)emb[(long)tok * D_MODEL_ + (tid % D_MODEL_)];
}

// causal depthwise conv (K=4) + bias + SiLU; reads xi half of xz (b,l,2*D_IN)
// writes f32 (for the scan) and bf16 (x_proj GEMM operand)
__global__ void mamba_conv_silu(const float* __restrict__ xz, const float* __restrict__ cw,
                                const float* __restrict__ cb,
                                float* __restrict__ xi, __bf16* __restrict__ xibf) {
  long tid = (long)blockIdx.x * blockDim.x + threadIdx.x;
  if (tid >= (long)M_TOK_ * D_IN_) return;
  int d = (int)(tid % D_IN_);
  int t = (int)((tid / D_IN_) % L_SEQ_);
  int b = (int)(tid / ((long)D_IN_ * L_SEQ_));
  float acc = cb[d];
  #pragma unroll
  for (int j = 0; j < K_CONV_; ++j) {
    int tt = t + j - (K_CONV_ - 1);
    if (tt >= 0) acc += cw[d * K_CONV_ + j] * xz[((long)b * L_SEQ_ + tt) * (2 * D_IN_) + d];
  }
  float s = acc / (1.f + __expf(-acc));  // silu
  xi[tid]   = s;
  xibf[tid] = (__bf16)s;
}

// selective scan: one lane per (b,d); 16 states in VGPRs; y overwrites dt in place
__global__ void mamba_scan(const float* __restrict__ xi, float* __restrict__ dty,
                           const float* __restrict__ xdbl, const float* __restrict__ Alog,
                           const float* __restrict__ Dsk) {
  const int d = blockIdx.x * blockDim.x + threadIdx.x;
  const int b = blockIdx.y;
  if (d >= D_IN_) return;
  float Ac[N_STATE_], hs[N_STATE_];
  #pragma unroll
  for (int n = 0; n < N_STATE_; ++n) { Ac[n] = -__expf(Alog[d * N_STATE_ + n]); hs[n] = 0.f; }
  const float dval = Dsk[d];
  const long base = (long)b * L_SEQ_;
  for (int t = 0; t < L_SEQ_; ++t) {
    const long idx = (base + t) * D_IN_ + d;
    const long bc  = (base + t) * XDBL_W_;
    if (t + 1 < L_SEQ_) __builtin_prefetch(xdbl + bc + XDBL_W_ + DT_RANK_, 0, 1);
    const float dt  = dty[idx];
    const float x   = xi[idx];
    const float dtx = dt * x;
    float y = 0.f;
    #pragma unroll
    for (int n = 0; n < N_STATE_; ++n) {
      const float Bt = xdbl[bc + DT_RANK_ + n];
      const float Ct = xdbl[bc + DT_RANK_ + N_STATE_ + n];
      hs[n] = __expf(dt * Ac[n]) * hs[n] + dtx * Bt;
      y += hs[n] * Ct;
    }
    dty[idx] = y + x * dval;
  }
}

// g = y * silu(z), z = second half of xz; bf16 output (out_proj GEMM operand)
__global__ void mamba_gate(const float* __restrict__ y, const float* __restrict__ xz,
                           __bf16* __restrict__ gbf) {
  long tid = (long)blockIdx.x * blockDim.x + threadIdx.x;
  if (tid >= (long)M_TOK_ * D_IN_) return;
  int d = (int)(tid % D_IN_);
  long row = tid / D_IN_;
  float z = xz[row * (2 * D_IN_) + D_IN_ + d];
  gbf[tid] = (__bf16)(y[tid] * (z / (1.f + __expf(-z))));
}

__global__ void mamba_pool(const float* __restrict__ h, float* __restrict__ pooled) {
  int tid = blockIdx.x * blockDim.x + threadIdx.x;
  if (tid >= B_SZ_ * D_MODEL_) return;
  int b = tid / D_MODEL_, dd = tid % D_MODEL_;
  float s = 0.f;
  for (int t = 0; t < L_SEQ_; ++t) s += h[((long)b * L_SEQ_ + t) * D_MODEL_ + dd];
  pooled[tid] = s * (1.f / (float)L_SEQ_);
}

__global__ void mamba_cls(const float* __restrict__ pooled, const float* __restrict__ cw,
                          const float* __restrict__ cb, float* __restrict__ out) {
  int tid = threadIdx.x;
  if (tid >= B_SZ_ * NC_) return;
  int b = tid / NC_, c = tid % NC_;
  float s = cb[c];
  for (int dd = 0; dd < D_MODEL_; ++dd) s += pooled[b * D_MODEL_ + dd] * cw[c * D_MODEL_ + dd];
  out[b * NC_ + c] = s;
}

// ---------------------------------------------------------------------------
static inline void cvt_launch(const float* src, __bf16* dst, long n, hipStream_t s) {
  mamba_cvt_bf16<<<(int)((n + 255) / 256), 256, 0, s>>>(src, dst, n);
}

extern "C" void kernel_launch(void* const* d_in, const int* in_sizes, int n_in,
                              void* d_out, int out_size, void* d_ws, size_t ws_size,
                              hipStream_t stream) {
  (void)in_sizes; (void)n_in; (void)out_size; (void)ws_size;
  const int*   tokens = (const int*)  d_in[0];
  const float* emb    = (const float*)d_in[1];
  const float* ipw    = (const float*)d_in[2];
  const float* cw     = (const float*)d_in[3];
  const float* cb     = (const float*)d_in[4];
  const float* xpw    = (const float*)d_in[5];
  const float* dtw    = (const float*)d_in[6];
  const float* dtb    = (const float*)d_in[7];
  const float* Alog   = (const float*)d_in[8];
  const float* Dsk    = (const float*)d_in[9];
  const float* opw    = (const float*)d_in[10];
  const float* clsw   = (const float*)d_in[11];
  const float* clsb   = (const float*)d_in[12];
  float* out = (float*)d_out;

  // ---- workspace carving: f32 region first, then bf16 region (64B-aligned)
  float* A      = (float*)d_ws;                              // h        (b,l,768)   f32
  float* Bxz    = A    + (size_t)M_TOK_ * D_MODEL_;          // xz       (b,l,3072)  f32
  float* C      = Bxz  + (size_t)M_TOK_ * 2 * D_IN_;         // xi       (b,l,1536)  f32
  float* Ddbl   = C    + (size_t)M_TOK_ * D_IN_;             // x_dbl    (b,l,80)    f32
  float* E      = Ddbl + (size_t)M_TOK_ * XDBL_W_;           // dt -> y  (b,l,1536)  f32
  float* pooled = E    + (size_t)M_TOK_ * D_IN_;             // (b,768)
  __bf16* Abf   = (__bf16*)(pooled + B_SZ_ * D_MODEL_);      // h        bf16
  __bf16* Cbf   = Abf  + (size_t)M_TOK_ * D_MODEL_;          // xi / g   bf16
  __bf16* Dbf   = Cbf  + (size_t)M_TOK_ * D_IN_;             // x_dbl    bf16
  __bf16* Wip   = Dbf  + (size_t)M_TOK_ * XDBL_W_;           // weights  bf16
  __bf16* Wxp   = Wip  + (size_t)(2 * D_IN_) * D_MODEL_;
  __bf16* Wdt   = Wxp  + (size_t)XDBL_W_ * D_IN_;
  __bf16* Wop   = Wdt  + (size_t)D_IN_ * DT_RANK_;

  const dim3 blk(128);
  mamba_embed<<<(M_TOK_ * D_MODEL_ + 255) / 256, 256, 0, stream>>>(tokens, emb, Abf);

  for (int layer = 0; layer < N_LAYERS_; ++layer) {
    const float* ipw_l  = ipw  + (size_t)layer * (2 * D_IN_) * D_MODEL_;
    const float* cw_l   = cw   + (size_t)layer * D_IN_ * K_CONV_;
    const float* cb_l   = cb   + (size_t)layer * D_IN_;
    const float* xpw_l  = xpw  + (size_t)layer * XDBL_W_ * D_IN_;
    const float* dtw_l  = dtw  + (size_t)layer * D_IN_ * DT_RANK_;
    const float* dtb_l  = dtb  + (size_t)layer * D_IN_;
    const float* Alog_l = Alog + (size_t)layer * D_IN_ * N_STATE_;
    const float* Dsk_l  = Dsk  + (size_t)layer * D_IN_;
    const float* opw_l  = opw  + (size_t)layer * D_MODEL_ * D_IN_;

    // stage this layer's weights as bf16 (once, not per M-tile)
    cvt_launch(ipw_l, Wip, (long)(2 * D_IN_) * D_MODEL_, stream);
    cvt_launch(xpw_l, Wxp, (long)XDBL_W_ * D_IN_, stream);
    cvt_launch(dtw_l, Wdt, (long)D_IN_ * DT_RANK_, stream);
    cvt_launch(opw_l, Wop, (long)D_MODEL_ * D_IN_, stream);

    // xz = h @ ipw.T                       (4096 x 768 -> 3072)
    mamba_gemm<<<dim3((2 * D_IN_) / 64, M_TOK_ / 64), blk, 0, stream>>>(
        Abf, Wip, nullptr, Bxz, nullptr, M_TOK_, D_MODEL_, D_MODEL_, 2 * D_IN_, 0);
    // xi = silu(causal_conv(x) + cb)  -> f32 + bf16
    mamba_conv_silu<<<(int)(((long)M_TOK_ * D_IN_ + 255) / 256), 256, 0, stream>>>(
        Bxz, cw_l, cb_l, C, Cbf);
    // x_dbl = xi @ xpw.T                   (4096 x 1536 -> 80), f32 + bf16 mirror
    mamba_gemm<<<dim3(2, M_TOK_ / 64), blk, 0, stream>>>(
        Cbf, Wxp, nullptr, Ddbl, Dbf, M_TOK_, D_IN_, D_IN_, XDBL_W_, 0);
    // dt = softplus(x_dbl[:, :48] @ dtw.T + dtb)   (K=48 tail zero-padded, lda=80)
    mamba_gemm<<<dim3(D_IN_ / 64, M_TOK_ / 64), blk, 0, stream>>>(
        Dbf, Wdt, dtb_l, E, nullptr, M_TOK_, DT_RANK_, XDBL_W_, D_IN_, 1);
    // selective scan: y (in place over dt)
    mamba_scan<<<dim3(D_IN_ / 256, B_SZ_), 256, 0, stream>>>(C, E, Ddbl, Alog_l, Dsk_l);
    // g = y * silu(z) -> bf16 operand
    mamba_gate<<<(int)(((long)M_TOK_ * D_IN_ + 255) / 256), 256, 0, stream>>>(E, Bxz, Cbf);
    // h = g @ opw.T                        (4096 x 1536 -> 768), f32 (pool) + bf16 (next layer)
    mamba_gemm<<<dim3(D_MODEL_ / 64, M_TOK_ / 64), blk, 0, stream>>>(
        Cbf, Wop, nullptr, A, Abf, M_TOK_, D_IN_, D_IN_, D_MODEL_, 0);
  }

  mamba_pool<<<(B_SZ_ * D_MODEL_ + 255) / 256, 256, 0, stream>>>(A, pooled);
  mamba_cls<<<1, 32, 0, stream>>>(pooled, clsw, clsb, out);
}
